// CPUMamba_86337432584510
// MI455X (gfx1250) — compile-verified, matrix operand order checked by reading, not run on hardware
//
#include <hip/hip_runtime.h>
#include <hip/hip_bf16.h>

// ---------------------------------------------------------------------------
// Mamba forward for MI455X (gfx1250).
// bf16 WMMA GEMMs: A via global b128 loads, B weight tiles staged into LDS
// with async-to-LDS DMA (ASYNCcnt) double buffering, fragments via ds_load.
// Fused depthwise conv + SiLU, fused selective scan + gating.
// ---------------------------------------------------------------------------

#define D_MODEL 1024
#define D_STATE 16
#define D_CONV  4
#define D_INNER 2048
#define DT_RANK 64
#define BB      2
#define TT      2048
#define M_TOK   (BB * TT)          // 4096 token rows

typedef __attribute__((ext_vector_type(16))) __bf16 v16bf;
typedef __attribute__((ext_vector_type(8)))  __bf16 v8bf;
typedef __attribute__((ext_vector_type(8)))  float  v8f;

// fp32 -> bf16 round-to-nearest-even (used only in elementwise kernels)
static __device__ __forceinline__ __bf16 f2bf(float f) {
  union { float f; unsigned u; } x; x.f = f;
  unsigned r = x.u + 0x7FFFu + ((x.u >> 16) & 1u);
  unsigned short h = (unsigned short)(r >> 16);
  __bf16 b;
  __builtin_memcpy(&b, &h, 2);
  return b;
}

static __device__ __forceinline__ float siluf(float x) {
  return x / (1.0f + __expf(-x));
}
static __device__ __forceinline__ float softplusf(float x) {
  return (x > 20.0f) ? x : logf(1.0f + __expf(x));
}

// Combine two 16-byte fragment halves into one v16bf operand.
union Frag16 {
  v16bf v;
  v8bf  h[2];
};

// Async DMA: global (16B) -> LDS, tracked by ASYNCcnt.
static __device__ __forceinline__ void async_g2l_b128(const void* gptr, void* lptr) {
  const unsigned lds_off = (unsigned)(size_t)lptr;           // flat->LDS: low 32 bits
  const unsigned long long ga = (unsigned long long)(size_t)gptr;
  asm volatile("global_load_async_to_lds_b128 %0, %1, off"
               :: "v"(lds_off), "v"(ga) : "memory");
}

template <int N>
static __device__ __forceinline__ void wait_asynccnt() {
  asm volatile("s_wait_asynccnt %0" :: "i"(N) : "memory");
}

// ---------------------------------------------------------------------------
// Generic bf16 GEMM:  C[m,n] = act( sum_k A[m,k] * Bw[n,k] + bias[n] )
//   A:  (M x K) row-major bf16, leading dim lda   (activations)
//   Bw: (N x K) row-major bf16, leading dim ldb   (weights -> B^T in math)
// 4 waves / block stacked in M; each wave computes a 16 x (16*NT) tile with
// v_wmma_f32_16x16x32_bf16.  B tile (16*NT rows x 32 k, bf16) is staged into
// LDS by async DMA, double-buffered; all 4 waves share it.  Last k-chunk is
// peeled so the steady-state loop is branch-free.
// Requires M % 64 == 0, N % (16*NT) == 0, K % 32 == 0, NT in {2,4,8}.
// ACT: 0 = none, 1 = bias + softplus.
// ---------------------------------------------------------------------------
template <int NT, int ACT>
__global__ __launch_bounds__(128)
void gemm_wmma_bf16(const __bf16* __restrict__ A, int lda,
                    const __bf16* __restrict__ Bw, int ldb,
                    float* __restrict__ C, int ldc,
                    int K, const float* __restrict__ bias) {
  constexpr int ROWS = 16 * NT;          // B-tile rows per block
  constexpr int SEGS = ROWS * 4;         // 16B segments per tile (row = 64B)
  constexpr int ISS  = SEGS / 128;       // async issues per wave per stage

  __shared__ __align__(16) unsigned char bbuf[2][ROWS * 64];

  const int tid  = threadIdx.x;
  const int lane = tid & 31;
  const int wave = tid >> 5;
  const int lid  = lane & 15;    // row-within-tile (A) / col-within-tile (B)
  const int hsel = lane >> 4;    // half-select: K phase
  const int mrow = blockIdx.y * 64 + wave * 16 + lid;   // A row read by lane
  const int n0   = blockIdx.x * ROWS;

  const __bf16* arow = A + (size_t)mrow * lda + hsel * 8;

  // Stage one 32-k B tile (rows n0..n0+ROWS-1) into bbuf[buf] via async DMA.
  auto stage = [&](int buf, int kb) {
#pragma unroll
    for (int it = 0; it < ISS; ++it) {
      const int s   = tid + it * 128;
      const int row = s >> 2;
      const int seg = s & 3;
      const __bf16* g = Bw + (size_t)(n0 + row) * ldb + kb + seg * 8;
      async_g2l_b128(g, &bbuf[buf][row * 64 + seg * 16]);
    }
  };

  v8f acc[NT] = {};

  // One k-chunk of compute from LDS buffer `buf`.
  auto do_chunk = [&](int kb, int buf) {
    Frag16 af;
    af.h[0] = *(const v8bf*)(arow + kb);
    af.h[1] = *(const v8bf*)(arow + kb + 16);
#pragma unroll
    for (int j = 0; j < NT; ++j) {
      const unsigned char* bp = &bbuf[buf][(16 * j + lid) * 64 + hsel * 16];
      Frag16 bf;
      bf.h[0] = *(const v8bf*)(bp);        // ds_load_b128
      bf.h[1] = *(const v8bf*)(bp + 32);   // ds_load_b128
      acc[j] = __builtin_amdgcn_wmma_f32_16x16x32_bf16(
          false, af.v, false, bf.v, (short)0, acc[j], false, false);
    }
  };

  stage(0, 0);
  int cur = 0;
  int kb = 0;
  for (; kb + 32 < K; kb += 32) {        // steady state: branch-free body
    stage(cur ^ 1, kb + 32);
    wait_asynccnt<ISS>();                // just-issued may remain; cur buf done
    __syncthreads();                     // buf[cur] visible to all 4 waves
    __builtin_prefetch(arow + kb + 32, 0, 3);
    do_chunk(kb, cur);
    __syncthreads();                     // drain readers before re-staging
    cur ^= 1;
  }
  // peeled last chunk
  wait_asynccnt<0>();
  __syncthreads();
  do_chunk(kb, cur);

  // Store: VGPR e of C/D holds row (e + 8*hsel), col lid within the 16x16 tile.
#pragma unroll
  for (int j = 0; j < NT; ++j) {
    const int n = n0 + 16 * j + lid;
#pragma unroll
    for (int e = 0; e < 8; ++e) {
      const int row = blockIdx.y * 64 + wave * 16 + e + 8 * hsel;
      float v = acc[j][e];
      if (ACT == 1) v = softplusf(v + bias[n]);
      C[(size_t)row * ldc + n] = v;
    }
  }
}

// ---------------------------------------------------------------------------
// Elementwise fp32 -> bf16 (weights / x, once per call)
// ---------------------------------------------------------------------------
__global__ __launch_bounds__(256)
void f32_to_bf16_kernel(const float* __restrict__ src, __bf16* __restrict__ dst, int n) {
  const int i = blockIdx.x * blockDim.x + threadIdx.x;
  if (i < n) dst[i] = f2bf(src[i]);
}

// Extract dt_lo (first 64 cols of x_dbl, ld 96) into compact bf16 (ld 64)
__global__ __launch_bounds__(256)
void extract_dtlo_kernel(const float* __restrict__ x_dbl, __bf16* __restrict__ dst) {
  const int i = blockIdx.x * blockDim.x + threadIdx.x;   // over M_TOK * 64
  const int r = i >> 6;
  const int c = i & 63;
  dst[i] = f2bf(x_dbl[(size_t)r * 96 + c]);
}

// ---------------------------------------------------------------------------
// Depthwise causal conv1d (width 4) + SiLU.  Writes fp32 (scan) + bf16 (GEMM).
// ---------------------------------------------------------------------------
__global__ __launch_bounds__(256)
void conv_silu_kernel(const float* __restrict__ xz,
                      const float* __restrict__ conv_w,
                      const float* __restrict__ conv_b,
                      float* __restrict__ xc,
                      __bf16* __restrict__ xc_bf) {
  const int idx = blockIdx.x * blockDim.x + threadIdx.x;   // over B*T*D_INNER
  const int d = idx % D_INNER;
  const int t = (idx / D_INNER) % TT;
  const int b = idx / (D_INNER * TT);
  float acc = conv_b[d];
#pragma unroll
  for (int j = 0; j < D_CONV; ++j) {
    const int tt = t - (D_CONV - 1) + j;
    if (tt >= 0)
      acc += xz[((size_t)(b * TT + tt)) * (2 * D_INNER) + d] * conv_w[d * D_CONV + j];
  }
  const float v = siluf(acc);
  xc[(size_t)idx] = v;
  xc_bf[(size_t)idx] = f2bf(v);
}

// ---------------------------------------------------------------------------
// Selective scan.  One thread per (b, d) row; 16-element state in registers.
// B_t / C_t vectors staged via LDS in 32-timestep chunks.
// Fuses  y = (scan + D*x) * silu(z)  and writes bf16 for the out_proj GEMM.
// ---------------------------------------------------------------------------
#define TCHUNK 32
__global__ __launch_bounds__(256)
void scan_kernel(const float* __restrict__ xz,      // for z (ld 2*D_INNER)
                 const float* __restrict__ xc,      // (B*T, D_INNER) fp32
                 const float* __restrict__ x_dbl,   // (B*T, 96): [dt_lo | B | C]
                 const float* __restrict__ dt,      // (B*T, D_INNER) fp32
                 __bf16* __restrict__ yg_bf,        // out: gated y, bf16
                 const float* __restrict__ A_log,   // (D_INNER, D_STATE)
                 const float* __restrict__ Dp) {    // (D_INNER)
  const int d = blockIdx.x * blockDim.x + threadIdx.x;   // 0..D_INNER-1
  const int b = blockIdx.y;

  float Aneg[D_STATE];
#pragma unroll
  for (int s = 0; s < D_STATE; ++s)
    Aneg[s] = -__expf(A_log[d * D_STATE + s]);

  float h[D_STATE];
#pragma unroll
  for (int s = 0; s < D_STATE; ++s) h[s] = 0.0f;

  const float dval = Dp[d];

  __shared__ float sB[TCHUNK][D_STATE];
  __shared__ float sC[TCHUNK][D_STATE];

  for (int tc = 0; tc < TT; tc += TCHUNK) {
    for (int i = threadIdx.x; i < TCHUNK * 32; i += blockDim.x) {
      const int r = i >> 5;
      const int j = i & 31;
      const float v = x_dbl[((size_t)(b * TT + tc + r)) * 96 + DT_RANK + j];
      if (j < D_STATE) sB[r][j] = v; else sC[r][j - D_STATE] = v;
    }
    __syncthreads();

    for (int r = 0; r < TCHUNK; ++r) {
      const size_t rowi = (size_t)(b * TT + tc + r);
      const float dtv = dt[rowi * D_INNER + d];
      const float xv  = xc[rowi * D_INNER + d];
      float y = 0.0f;
#pragma unroll
      for (int s = 0; s < D_STATE; ++s) {
        const float dA = __expf(dtv * Aneg[s]);
        h[s] = dA * h[s] + dtv * sB[r][s] * xv;
        y += h[s] * sC[r][s];
      }
      const float zv = xz[rowi * (2 * D_INNER) + D_INNER + d];
      yg_bf[rowi * D_INNER + d] = f2bf((y + dval * xv) * siluf(zv));
    }
    __syncthreads();
  }
}

// ---------------------------------------------------------------------------
// Host-side orchestration
// ---------------------------------------------------------------------------
extern "C" void kernel_launch(void* const* d_in, const int* in_sizes, int n_in,
                              void* d_out, int out_size, void* d_ws, size_t ws_size,
                              hipStream_t stream) {
  const float* x         = (const float*)d_in[0];
  const float* in_proj_w = (const float*)d_in[1];
  const float* conv_w    = (const float*)d_in[2];
  const float* conv_b    = (const float*)d_in[3];
  const float* x_proj_w  = (const float*)d_in[4];
  const float* dt_proj_w = (const float*)d_in[5];
  const float* dt_proj_b = (const float*)d_in[6];
  const float* A_log     = (const float*)d_in[7];
  const float* Dp        = (const float*)d_in[8];
  const float* out_proj_w= (const float*)d_in[9];
  float* out = (float*)d_out;

  // ---- workspace layout (all offsets 16B-aligned) ----
  char* ws = (char*)d_ws;
  size_t ofs = 0;
  auto alloc_f32 = [&](size_t n) { float*  p = (float*) (ws + ofs); ofs += n * 4; return p; };
  auto alloc_bf  = [&](size_t n) { __bf16* p = (__bf16*)(ws + ofs); ofs += n * 2; return p; };

  float*  xz      = alloc_f32((size_t)M_TOK * 2 * D_INNER);  // 4096 x 4096
  float*  xc      = alloc_f32((size_t)M_TOK * D_INNER);      // 4096 x 2048
  float*  x_dbl   = alloc_f32((size_t)M_TOK * 96);           // 4096 x 96
  float*  dt      = alloc_f32((size_t)M_TOK * D_INNER);      // 4096 x 2048
  __bf16* xb      = alloc_bf((size_t)M_TOK * D_MODEL);       // x in bf16
  __bf16* w_in    = alloc_bf((size_t)2 * D_INNER * D_MODEL);
  __bf16* w_x     = alloc_bf((size_t)96 * D_INNER);
  __bf16* w_dt    = alloc_bf((size_t)D_INNER * DT_RANK);
  __bf16* w_out   = alloc_bf((size_t)D_MODEL * D_INNER);
  __bf16* xc_bf   = alloc_bf((size_t)M_TOK * D_INNER);
  __bf16* dtlo_bf = alloc_bf((size_t)M_TOK * DT_RANK);
  __bf16* yg_bf   = alloc_bf((size_t)M_TOK * D_INNER);

  auto cvt = [&](const float* s, __bf16* p, int n) {
    f32_to_bf16_kernel<<<(n + 255) / 256, 256, 0, stream>>>(s, p, n);
  };

  // 0) one-shot bf16 conversions (weights + x)
  cvt(x,          xb,    M_TOK * D_MODEL);
  cvt(in_proj_w,  w_in,  2 * D_INNER * D_MODEL);
  cvt(x_proj_w,   w_x,   96 * D_INNER);
  cvt(dt_proj_w,  w_dt,  D_INNER * DT_RANK);
  cvt(out_proj_w, w_out, D_MODEL * D_INNER);

  // 1) in_proj: xz = x @ in_proj_w^T     (4096 x 4096, K=1024)
  gemm_wmma_bf16<8, 0><<<dim3((2 * D_INNER) / 128, M_TOK / 64), 128, 0, stream>>>(
      xb, D_MODEL, w_in, D_MODEL, xz, 2 * D_INNER, D_MODEL, nullptr);

  // 2) depthwise causal conv + SiLU -> xc (fp32 + bf16)
  conv_silu_kernel<<<(M_TOK * D_INNER) / 256, 256, 0, stream>>>(
      xz, conv_w, conv_b, xc, xc_bf);

  // 3) x_proj: x_dbl = xc @ x_proj_w^T   (4096 x 96, K=2048)
  gemm_wmma_bf16<2, 0><<<dim3(96 / 32, M_TOK / 64), 128, 0, stream>>>(
      xc_bf, D_INNER, w_x, D_INNER, x_dbl, 96, D_INNER, nullptr);

  // 3b) compact bf16 dt_lo
  extract_dtlo_kernel<<<(M_TOK * DT_RANK) / 256, 256, 0, stream>>>(x_dbl, dtlo_bf);

  // 4) dt_proj + bias + softplus: dt = softplus(dt_lo @ dt_proj_w^T + b)
  gemm_wmma_bf16<8, 1><<<dim3(D_INNER / 128, M_TOK / 64), 128, 0, stream>>>(
      dtlo_bf, DT_RANK, w_dt, DT_RANK, dt, D_INNER, DT_RANK, dt_proj_b);

  // 5) selective scan + D*x + SiLU(z) gate -> yg (bf16)
  scan_kernel<<<dim3(D_INNER / 256, BB), 256, 0, stream>>>(
      xz, xc, x_dbl, dt, yg_bf, A_log, Dp);

  // 6) out_proj: out = y @ out_proj_w^T  (4096 x 1024, K=2048)
  gemm_wmma_bf16<8, 0><<<dim3(D_MODEL / 128, M_TOK / 64), 128, 0, stream>>>(
      yg_bf, D_INNER, w_out, D_INNER, out, D_MODEL, D_INNER, nullptr);
}
